// NetHY_16853451669863
// MI455X (gfx1250) — matrix operand order repaired and checked by reading, not compiled, not run on hardware
//
#include <hip/hip_runtime.h>
#include <math.h>

// Problem constants (match reference)
#define NN    4096
#define FIN   4096
#define HID   4096
#define CODE  64
#define TK    8

// ---------------------------------------------------------------------------
// bf16 helpers + WMMA vector types
// ---------------------------------------------------------------------------
typedef __attribute__((ext_vector_type(16))) __bf16 v16bf;
typedef __attribute__((ext_vector_type(8)))  __bf16 v8bf;
typedef __attribute__((ext_vector_type(8)))  float  v8f;

__device__ __forceinline__ unsigned short f2bf(float f) {
    unsigned u = __float_as_uint(f);
    unsigned r = (u >> 16) & 1u;          // round-to-nearest-even
    u += 0x7fffu + r;
    return (unsigned short)(u >> 16);
}

__device__ __forceinline__ unsigned pack2bf(float lo, float hi) {
    return (unsigned)f2bf(lo) | ((unsigned)f2bf(hi) << 16);
}

// ---------------------------------------------------------------------------
// 1) Top-K (K=8) per row of S. One block per row; per-thread sorted top-8,
//    then pairwise log2(256) merges in LDS. Ties: smaller index wins (jax).
// ---------------------------------------------------------------------------
__global__ __launch_bounds__(256) void topk_kernel(const float* __restrict__ S,
                                                   int* __restrict__ members) {
    __shared__ float sv[256 * TK];
    __shared__ int   si[256 * TK];
    const int row = blockIdx.x;
    const int tid = threadIdx.x;
    const float* r = S + (size_t)row * NN;

    float v[TK]; int id[TK];
    for (int i = 0; i < TK; ++i) { v[i] = -INFINITY; id[i] = 0x7fffffff; }

    for (int c = tid; c < NN; c += 256) {
        float x = r[c];
        if (x > v[TK - 1] || (x == v[TK - 1] && c < id[TK - 1])) {
            int j = TK - 1;
            while (j > 0 && (x > v[j - 1] || (x == v[j - 1] && c < id[j - 1]))) {
                v[j] = v[j - 1]; id[j] = id[j - 1]; --j;
            }
            v[j] = x; id[j] = c;
        }
    }
    for (int i = 0; i < TK; ++i) { sv[tid * TK + i] = v[i]; si[tid * TK + i] = id[i]; }
    __syncthreads();

    for (int s = 128; s > 0; s >>= 1) {
        if (tid < s) {
            float* A  = &sv[tid * TK];       int* AI = &si[tid * TK];
            float* B  = &sv[(tid + s) * TK]; int* BI = &si[(tid + s) * TK];
            float ov[TK]; int oi[TK];
            int pa = 0, pb = 0;
            for (int o = 0; o < TK; ++o) {
                bool takeA = (A[pa] > B[pb]) || (A[pa] == B[pb] && AI[pa] < BI[pb]);
                if (takeA) { ov[o] = A[pa]; oi[o] = AI[pa]; ++pa; }
                else       { ov[o] = B[pb]; oi[o] = BI[pb]; ++pb; }
            }
            for (int o = 0; o < TK; ++o) { A[o] = ov[o]; AI[o] = oi[o]; }
        }
        __syncthreads();
    }
    if (tid < TK) members[row * TK + tid] = si[tid];
}

// ---------------------------------------------------------------------------
// 2) CSR build: degree count -> block scan -> fill -> per-node sort
// ---------------------------------------------------------------------------
__global__ void count_deg(const int* __restrict__ members, int* __restrict__ deg) {
    int idx = blockIdx.x * blockDim.x + threadIdx.x;     // 0..N*TK-1
    atomicAdd(&deg[members[idx]], 1);
}

__global__ __launch_bounds__(1024) void scan_kernel(const int* __restrict__ deg,
                                                    int* __restrict__ offsets) {
    const int PER = NN / 1024;                            // 4
    __shared__ int tsum[1024];
    int tid  = threadIdx.x;
    int base = tid * PER;
    int local[PER];
    int s = 0;
    for (int i = 0; i < PER; ++i) { local[i] = s; s += deg[base + i]; }
    tsum[tid] = s;
    __syncthreads();
    for (int off = 1; off < 1024; off <<= 1) {
        int v   = tsum[tid];
        int add = (tid >= off) ? tsum[tid - off] : 0;
        __syncthreads();
        tsum[tid] = v + add;
        __syncthreads();
    }
    int prev = (tid == 0) ? 0 : tsum[tid - 1];
    for (int i = 0; i < PER; ++i) offsets[base + i] = prev + local[i];
    if (tid == 1023) offsets[NN] = tsum[1023];
}

__global__ void fill_lists(const int* __restrict__ members, const int* __restrict__ offsets,
                           int* __restrict__ cursor, int* __restrict__ edge_list) {
    int idx = blockIdx.x * blockDim.x + threadIdx.x;     // 0..N*TK-1
    int e = idx >> 3;
    int m = members[idx];
    int pos = atomicAdd(&cursor[m], 1);
    edge_list[offsets[m] + pos] = e;
}

// Sort each node's incidence list (avg length 8) => deterministic sum order.
__global__ void sort_lists(const int* __restrict__ deg, const int* __restrict__ offsets,
                           int* __restrict__ edge_list) {
    if (threadIdx.x != 0) return;
    int m = blockIdx.x;
    int s = offsets[m];
    int c = deg[m];
    for (int i = 1; i < c; ++i) {
        int v = edge_list[s + i];
        int j = i - 1;
        while (j >= 0 && edge_list[s + j] > v) { edge_list[s + j + 1] = edge_list[s + j]; --j; }
        edge_list[s + j + 1] = v;
    }
}

// ---------------------------------------------------------------------------
// 3) bf16 WMMA GEMM: C[M,N] = A[M,K] * B[K,N]; f32 in global, bf16 in LDS.
//    256 threads = 8 wave32s; wave tile (BM/WR) x (BN/WC); BK = 32.
//    Double-buffered LDS: next tile's global loads overlap current WMMAs.
// ---------------------------------------------------------------------------
template<int BM, int BN, int WR, int WC>
__global__ __launch_bounds__(256) void gemm_bf16_wmma(const float* __restrict__ A,
                                                      const float* __restrict__ B,
                                                      float* __restrict__ C,
                                                      int M, int N, int Kd) {
    constexpr int BK  = 32;
    constexpr int LDA = BK + 8;        // 40 halves = 80B row stride (16B aligned)
    constexpr int LDB = BK + 8;
    constexpr int WM  = BM / WR;
    constexpr int WN  = BN / WC;
    constexpr int FM  = WM / 16;
    constexpr int FN  = WN / 16;
    __shared__ unsigned short ldsA[2][BM * LDA];
    __shared__ unsigned short ldsB[2][BN * LDB];   // stored transposed: [n][k]

    const int tid      = threadIdx.x;
    const int mBlock   = blockIdx.y * BM;
    const int nBlock   = blockIdx.x * BN;
    const int wave     = tid >> 5;
    const int lane     = tid & 31;
    const int laneRow  = lane & 15;
    const int laneHalf = lane >> 4;
    const int waveM    = (wave % WR) * WM;
    const int waveN    = (wave / WR) * WN;

    // Tile loader: global f32 -> bf16 -> LDS (A row-major [m][k], B transposed [n][k])
    auto loadTile = [&](int kt, int buf) {
        // ---- A tile: 4 float4 per thread, packed pair stores (8B, aligned) ----
        constexpr int A_ITERS = (BM * BK) / (256 * 4);
        for (int i = 0; i < A_ITERS; ++i) {
            int flat = tid + i * 256;                 // float4 index
            int row  = flat >> 3;                     // 8 float4 per row (BK=32)
            int cg   = flat & 7;
            const float4 vv = *(const float4*)(A + (size_t)(mBlock + row) * Kd + kt + cg * 4);
            uint2 t; t.x = pack2bf(vv.x, vv.y); t.y = pack2bf(vv.z, vv.w);
            *(uint2*)&ldsA[buf][row * LDA + cg * 4] = t;
            if (kt + BK < Kd)                          // prefetch one tile ahead
                __builtin_prefetch(A + (size_t)(mBlock + row) * Kd + kt + BK + cg * 4, 0, 1);
        }
        // ---- B tile: two adjacent k-rows per unit -> packed b32 stores ----
        constexpr int B_ITERS = (BN * BK) / (256 * 8);
        for (int i = 0; i < B_ITERS; ++i) {
            int unit = tid + i * 256;
            int kp   = unit / (BN / 4);               // k-pair index (0..15)
            int cg   = unit % (BN / 4);               // n group of 4
            const float4 v0 = *(const float4*)(B + (size_t)(kt + 2 * kp + 0) * N + nBlock + cg * 4);
            const float4 v1 = *(const float4*)(B + (size_t)(kt + 2 * kp + 1) * N + nBlock + cg * 4);
            *(unsigned*)&ldsB[buf][(cg * 4 + 0) * LDB + 2 * kp] = pack2bf(v0.x, v1.x);
            *(unsigned*)&ldsB[buf][(cg * 4 + 1) * LDB + 2 * kp] = pack2bf(v0.y, v1.y);
            *(unsigned*)&ldsB[buf][(cg * 4 + 2) * LDB + 2 * kp] = pack2bf(v0.z, v1.z);
            *(unsigned*)&ldsB[buf][(cg * 4 + 3) * LDB + 2 * kp] = pack2bf(v0.w, v1.w);
        }
    };

    v8f acc[FM][FN];
    for (int i = 0; i < FM; ++i)
        for (int j = 0; j < FN; ++j)
            for (int r = 0; r < 8; ++r) acc[i][j][r] = 0.0f;

    loadTile(0, 0);
    __syncthreads();

    int cur = 0;
    for (int k0 = 0; k0 < Kd; k0 += BK) {
        if (k0 + BK < Kd) loadTile(k0 + BK, cur ^ 1);   // overlap with WMMAs below

        union FragU { v16bf v; v8bf h[2]; };
        FragU af[FM], bfr[FN];
        // A 16x32 bf16 layout: lanes<16 K{0..7,16..23}, lanes>=16 K{8..15,24..31}
        for (int i = 0; i < FM; ++i) {
            int base = (waveM + 16 * i + laneRow) * LDA + laneHalf * 8;
            af[i].h[0] = *(const v8bf*)&ldsA[cur][base];
            af[i].h[1] = *(const v8bf*)&ldsA[cur][base + 16];
        }
        // B 32x16 bf16 layout: lane n holds column n, lanes<16 K{0..7,16..23}, lanes>=16 K{8..15,24..31}
        for (int j = 0; j < FN; ++j) {
            int base = (waveN + 16 * j + laneRow) * LDB + laneHalf * 8;
            bfr[j].h[0] = *(const v8bf*)&ldsB[cur][base];
            bfr[j].h[1] = *(const v8bf*)&ldsB[cur][base + 16];
        }
        for (int i = 0; i < FM; ++i)
            for (int j = 0; j < FN; ++j)
                acc[i][j] = __builtin_amdgcn_wmma_f32_16x16x32_bf16(
                    false, af[i].v, false, bfr[j].v, (short)0, acc[i][j], false, false);

        __syncthreads();   // reads of buf[cur] done; writes of buf[cur^1] visible
        cur ^= 1;
    }

    // C/D layout: VGPR r -> M = r + 8*(lane>=16), N = lane&15
    for (int i = 0; i < FM; ++i)
        for (int j = 0; j < FN; ++j)
            for (int r = 0; r < 8; ++r) {
                int gr = mBlock + waveM + 16 * i + r + 8 * laneHalf;
                int gc = nBlock + waveN + 16 * j + laneRow;
                C[(size_t)gr * N + gc] = acc[i][j][r];
            }
}

// ---------------------------------------------------------------------------
// 4) edge_feat[e][:] = mean_k xw[members[e][k]][:]   (block per edge)
// ---------------------------------------------------------------------------
__global__ void edge_mean(const float* __restrict__ xw, const int* __restrict__ members,
                          float* __restrict__ ef, int F) {
    const int e = blockIdx.x;
    int m[TK];
    for (int k = 0; k < TK; ++k) m[k] = members[e * TK + k];
    for (int f4 = threadIdx.x; f4 < (F >> 2); f4 += blockDim.x) {
        float4 s = make_float4(0.f, 0.f, 0.f, 0.f);
        for (int k = 0; k < TK; ++k) {
            float4 v = ((const float4*)(xw + (size_t)m[k] * F))[f4];
            s.x += v.x; s.y += v.y; s.z += v.z; s.w += v.w;
        }
        float4 o; o.x = s.x * 0.125f; o.y = s.y * 0.125f; o.z = s.z * 0.125f; o.w = s.w * 0.125f;
        ((float4*)(ef + (size_t)e * F))[f4] = o;
    }
}

// ---------------------------------------------------------------------------
// 5) agg[m][:] = inv_deg * sum_{e in list(m)} ef[e][:] + bias; activation.
//    ACT: 0 = relu, 1 = tanh. (block per node, CSR gather, sorted => determ.)
// ---------------------------------------------------------------------------
template<int ACT>
__global__ void agg_kernel(const float* __restrict__ ef, const int* __restrict__ deg,
                           const int* __restrict__ offsets, const int* __restrict__ edge_list,
                           const float* __restrict__ bias, float* __restrict__ out, int F) {
    const int m  = blockIdx.x;
    const int c  = deg[m];
    const int s0 = offsets[m];
    const float inv = (c > 0) ? (1.0f / (float)c) : 0.0f;
    for (int f4 = threadIdx.x; f4 < (F >> 2); f4 += blockDim.x) {
        float4 a = make_float4(0.f, 0.f, 0.f, 0.f);
        for (int t = 0; t < c; ++t) {
            int e = edge_list[s0 + t];
            float4 v = ((const float4*)(ef + (size_t)e * F))[f4];
            a.x += v.x; a.y += v.y; a.z += v.z; a.w += v.w;
        }
        float4 b = ((const float4*)bias)[f4];
        float4 r;
        r.x = a.x * inv + b.x; r.y = a.y * inv + b.y;
        r.z = a.z * inv + b.z; r.w = a.w * inv + b.w;
        if (ACT == 0) {
            r.x = fmaxf(r.x, 0.f); r.y = fmaxf(r.y, 0.f);
            r.z = fmaxf(r.z, 0.f); r.w = fmaxf(r.w, 0.f);
        } else {
            r.x = tanhf(r.x); r.y = tanhf(r.y); r.z = tanhf(r.z); r.w = tanhf(r.w);
        }
        ((float4*)(out + (size_t)m * F))[f4] = r;
    }
}

// ---------------------------------------------------------------------------
// Orchestration
// ---------------------------------------------------------------------------
static inline char* take(char*& p, size_t bytes) {
    char* r = p;
    p += (bytes + 255) & ~(size_t)255;
    return r;
}

extern "C" void kernel_launch(void* const* d_in, const int* in_sizes, int n_in,
                              void* d_out, int out_size, void* d_ws, size_t ws_size,
                              hipStream_t stream) {
    (void)in_sizes; (void)n_in; (void)out_size; (void)ws_size;

    const float* x  = (const float*)d_in[0];   // [N, FIN]
    const float* S  = (const float*)d_in[1];   // [N, N]
    const float* w1 = (const float*)d_in[2];   // [FIN, HID]
    const float* b1 = (const float*)d_in[3];   // [HID]
    const float* w2 = (const float*)d_in[4];   // [HID, CODE]
    const float* b2 = (const float*)d_in[5];   // [CODE]
    float* out = (float*)d_out;                // [N, CODE]

    char* p = (char*)d_ws;
    int*   members   = (int*)  take(p, (size_t)NN * TK * 4);
    int*   deg       = (int*)  take(p, (size_t)NN * 4);
    int*   offsets   = (int*)  take(p, (size_t)(NN + 1) * 4);
    int*   cursor    = (int*)  take(p, (size_t)NN * 4);
    int*   edge_list = (int*)  take(p, (size_t)NN * TK * 4);
    float* xw        = (float*)take(p, (size_t)NN * HID * 4);   // also reused as h
    float* ef        = (float*)take(p, (size_t)NN * HID * 4);
    float* xw2       = (float*)take(p, (size_t)NN * CODE * 4);
    float* ef2       = (float*)take(p, (size_t)NN * CODE * 4);
    float* h = xw;   // layer-1 output overwrites xw (xw dead after edge_mean)

    // --- hypergraph structure ---
    topk_kernel<<<NN, 256, 0, stream>>>(S, members);
    hipMemsetAsync(deg, 0, (size_t)NN * 4, stream);
    hipMemsetAsync(cursor, 0, (size_t)NN * 4, stream);
    count_deg<<<(NN * TK) / 256, 256, 0, stream>>>(members, deg);
    scan_kernel<<<1, 1024, 0, stream>>>(deg, offsets);
    fill_lists<<<(NN * TK) / 256, 256, 0, stream>>>(members, offsets, cursor, edge_list);
    sort_lists<<<NN, 32, 0, stream>>>(deg, offsets, edge_list);

    // --- layer 1: relu(D^-1 H B^-1 H^T (x @ w1) + b1) ---
    {
        dim3 grid(HID / 128, NN / 128);
        gemm_bf16_wmma<128, 128, 4, 2><<<grid, 256, 0, stream>>>(x, w1, xw, NN, HID, FIN);
    }
    edge_mean<<<NN, 256, 0, stream>>>(xw, members, ef, HID);
    agg_kernel<0><<<NN, 256, 0, stream>>>(ef, deg, offsets, edge_list, b1, h, HID);

    // --- layer 2: tanh(D^-1 H B^-1 H^T (h @ w2) + b2) ---
    {
        dim3 grid(CODE / 64, NN / 128);
        gemm_bf16_wmma<128, 64, 4, 2><<<grid, 256, 0, stream>>>(h, w2, xw2, NN, CODE, HID);
    }
    edge_mean<<<NN, 64, 0, stream>>>(xw2, members, ef2, CODE);
    agg_kernel<1><<<NN, 64, 0, stream>>>(ef2, deg, offsets, edge_list, b2, out, CODE);
}
